// PressureLSTM_16758962389168
// MI455X (gfx1250) — matrix operand order repaired and checked by reading, not compiled, not run on hardware
//
#include <hip/hip_runtime.h>

typedef __attribute__((ext_vector_type(16))) _Float16 v16h;
typedef __attribute__((ext_vector_type(8)))  _Float16 v8h;
typedef __attribute__((ext_vector_type(4)))  _Float16 v4h;
typedef __attribute__((ext_vector_type(8)))  float    v8f;
typedef __attribute__((ext_vector_type(4)))  float    v4f;

#define B_ 4096
#define T_ 512
#define F_ 32
#define H_ 32
#define GSTRIDE 20   // gatesT: padded float stride per column (bank-conflict free)
#define HSTRIDE 40   // hbuf: padded half stride per row (16B-aligned b128 loads)

// ---- fast activations: use CDNA5 V_TANH_F32 when available; never emit IEEE divide ----
#if __has_builtin(__builtin_amdgcn_tanhf)
__device__ __forceinline__ float ftanh_(float z) { return __builtin_amdgcn_tanhf(z); }
__device__ __forceinline__ float fsig(float z)   { return fmaf(0.5f, __builtin_amdgcn_tanhf(0.5f * z), 0.5f); }
#else
__device__ __forceinline__ float fsig(float z)   { return __builtin_amdgcn_rcpf(1.f + __expf(-z)); }
__device__ __forceinline__ float ftanh_(float z) { return fmaf(2.f, fsig(2.f * z), -1.f); }
#endif

__global__ __launch_bounds__(128) void lstm_fused_kernel(
    const float* __restrict__ x,     // [B, T, F]
    const float* __restrict__ W_ih,  // [4H, F]
    const float* __restrict__ W_hh,  // [4H, H]
    const float* __restrict__ b_ih,  // [4H]
    const float* __restrict__ b_hh,  // [4H]
    const float* __restrict__ W_fc,  // [1, H]
    const float* __restrict__ b_fc,  // [1]
    float* __restrict__ out)         // [B]
{
    __shared__ float    gatesT[128 * GSTRIDE];  // [gate_col][m] column-major, padded
    __shared__ _Float16 hbuf[16 * HSTRIDE];     // [row][k] f16, padded
    __shared__ float    redbuf[128];

    const int tid  = threadIdx.x;
    const int lane = tid & 31;
    const int wid  = tid >> 5;     // wave 0..3 -> gate group i,f,g,o
    const int lr   = lane & 15;    // 0..15
    const int lg   = lane >> 4;    // 0/1 (lane half-group)
    const int rowbase = blockIdx.x * 16;

    // ---- loop-invariant weight B-fragments (B[k][n] = W[n][k], K=32, N=16) ----
    // B layout used here (contiguous k, same order as A): lane L holds col n0+(L%16), K = (L/16)*16 + j, j=0..15
    const int n0a = wid * 32;
    const int n0b = n0a + 16;

    v16h bx0, bx1, bh0, bh1;
    {
        const float* p;
        p = W_ih + (size_t)(n0a + lr) * F_ + lg * 16;
        #pragma unroll
        for (int j = 0; j < 16; ++j) bx0[j] = (_Float16)p[j];
        p = W_ih + (size_t)(n0b + lr) * F_ + lg * 16;
        #pragma unroll
        for (int j = 0; j < 16; ++j) bx1[j] = (_Float16)p[j];
        p = W_hh + (size_t)(n0a + lr) * H_ + lg * 16;
        #pragma unroll
        for (int j = 0; j < 16; ++j) bh0[j] = (_Float16)p[j];
        p = W_hh + (size_t)(n0b + lr) * H_ + lg * 16;
        #pragma unroll
        for (int j = 0; j < 16; ++j) bh1[j] = (_Float16)p[j];
    }
    // C-fragment bias: element (m,n) depends only on n = n0 + (lane%16)
    const float bias0 = b_ih[n0a + lr] + b_hh[n0a + lr];
    const float bias1 = b_ih[n0b + lr] + b_hh[n0b + lr];

    // ---- h = 0 init ----
    for (int i = tid; i < 16 * HSTRIDE; i += 128) hbuf[i] = (_Float16)0.f;

    // elementwise ownership: row em (0..15), hidden cols ej..ej+3
    const int em = tid >> 3;
    const int ej = (tid & 7) * 4;
    float creg[4] = {0.f, 0.f, 0.f, 0.f};

    // A-fragment addressing: lane L -> row = rowbase + L%16, kbase = (L/16)*16
    const int  arow = rowbase + lr;
    const int  kb   = lg * 16;
    const float* xrow = x + (size_t)arow * T_ * F_ + kb;

    __syncthreads();

    for (int t = 0; t < T_; ++t) {
        // ---- A_x fragment from global (f32 -> f16) ----
        const float* xp = xrow + (size_t)t * F_;
        v4f f0 = *(const v4f*)(xp);
        v4f f1 = *(const v4f*)(xp + 4);
        v4f f2 = *(const v4f*)(xp + 8);
        v4f f3 = *(const v4f*)(xp + 12);
        if (t + 1 < T_) __builtin_prefetch(xp + F_, 0, 3);   // global_prefetch next step
        v16h ax;
        #pragma unroll
        for (int j = 0; j < 4; ++j) {
            ax[j]      = (_Float16)f0[j];
            ax[4 + j]  = (_Float16)f1[j];
            ax[8 + j]  = (_Float16)f2[j];
            ax[12 + j] = (_Float16)f3[j];
        }

        // ---- A_h fragment from LDS ----
        v8h h0 = *(const v8h*)(hbuf + lr * HSTRIDE + kb);
        v8h h1 = *(const v8h*)(hbuf + lr * HSTRIDE + kb + 8);
        v16h ah;
        #pragma unroll
        for (int j = 0; j < 8; ++j) { ah[j] = h0[j]; ah[8 + j] = h1[j]; }

        // ---- gates = x*Wih^T + h*Whh^T + b  (4 WMMAs per wave) ----
        v8f acc0, acc1;
        #pragma unroll
        for (int k = 0; k < 8; ++k) { acc0[k] = bias0; acc1[k] = bias1; }
        acc0 = __builtin_amdgcn_wmma_f32_16x16x32_f16(false, ax, false, bx0, (short)0, acc0, false, false);
        acc0 = __builtin_amdgcn_wmma_f32_16x16x32_f16(false, ah, false, bh0, (short)0, acc0, false, false);
        acc1 = __builtin_amdgcn_wmma_f32_16x16x32_f16(false, ax, false, bx1, (short)0, acc1, false, false);
        acc1 = __builtin_amdgcn_wmma_f32_16x16x32_f16(false, ah, false, bh1, (short)0, acc1, false, false);

        // ---- D tiles -> LDS column-major: elem k of D is (m = lg*8+k, n = n0+lr) ----
        {
            float* g0 = gatesT + (n0a + lr) * GSTRIDE + lg * 8;
            float* g1 = gatesT + (n0b + lr) * GSTRIDE + lg * 8;
            v4f s;
            #pragma unroll
            for (int k = 0; k < 4; ++k) s[k] = acc0[k];
            *(v4f*)(g0) = s;
            #pragma unroll
            for (int k = 0; k < 4; ++k) s[k] = acc0[4 + k];
            *(v4f*)(g0 + 4) = s;
            #pragma unroll
            for (int k = 0; k < 4; ++k) s[k] = acc1[k];
            *(v4f*)(g1) = s;
            #pragma unroll
            for (int k = 0; k < 4; ++k) s[k] = acc1[4 + k];
            *(v4f*)(g1 + 4) = s;
        }
        __syncthreads();

        // ---- LSTM cell elementwise; c persistent in registers ----
        v4h hh;
        #pragma unroll
        for (int q = 0; q < 4; ++q) {
            const int j = ej + q;
            float iv = gatesT[(j)*GSTRIDE + em];
            float fv = gatesT[(32 + j) * GSTRIDE + em];
            float gv = gatesT[(64 + j) * GSTRIDE + em];
            float ov = gatesT[(96 + j) * GSTRIDE + em];
            iv = fsig(iv);
            fv = fsig(fv);
            gv = ftanh_(gv);
            ov = fsig(ov);
            const float cn = fmaf(fv, creg[q], iv * gv);
            creg[q] = cn;
            hh[q] = (_Float16)(ov * ftanh_(cn));
        }
        *(v4h*)(hbuf + em * HSTRIDE + ej) = hh;
        __syncthreads();
    }

    // ---- final FC: out[row] = h_last . W_fc + b_fc ----
    float part = 0.f;
    #pragma unroll
    for (int q = 0; q < 4; ++q)
        part += (float)hbuf[em * HSTRIDE + ej + q] * W_fc[ej + q];
    redbuf[tid] = part;
    __syncthreads();
    if ((tid & 7) == 0) {
        float s = 0.f;
        #pragma unroll
        for (int q = 0; q < 8; ++q) s += redbuf[(em << 3) + q];
        out[rowbase + em] = s + b_fc[0];
    }
}

extern "C" void kernel_launch(void* const* d_in, const int* in_sizes, int n_in,
                              void* d_out, int out_size, void* d_ws, size_t ws_size,
                              hipStream_t stream) {
    const float* x    = (const float*)d_in[0];
    const float* W_ih = (const float*)d_in[1];
    const float* W_hh = (const float*)d_in[2];
    const float* b_ih = (const float*)d_in[3];
    const float* b_hh = (const float*)d_in[4];
    const float* W_fc = (const float*)d_in[5];
    const float* b_fc = (const float*)d_in[6];
    float* out = (float*)d_out;

    lstm_fused_kernel<<<B_ / 16, 128, 0, stream>>>(x, W_ih, W_hh, b_ih, b_hh, W_fc, b_fc, out);

    (void)in_sizes; (void)n_in; (void)out_size; (void)d_ws; (void)ws_size;
}
